// H_Attention_12429635355150
// MI455X (gfx1250) — matrix operand order repaired and checked by reading, not compiled, not run on hardware
//
#include <hip/hip_runtime.h>
#include <hip/hip_bf16.h>

// ---------------------------------------------------------------------------
// Axial (height) attention, MI455X / gfx1250, wave32 + WMMA bf16.
//   B=4, C=512, CQK=64, H=W=128
// Kernel 1: fused q/k/v 1x1-conv projection (x read once, bf16 WMMA)
// Kernel 2: per-(b,w) column attention: QK^T -> softmax -> P*V -> gamma*out+x
//   - WMMA fragments fetched as 2x ds_load_b128 (no scalar gathers)
//   - V tiles streamed with GLOBAL_LOAD_ASYNC_TO_LDS_B128 (ASYNCcnt),
//     double-buffered so loads overlap the WMMA stream.
// ---------------------------------------------------------------------------

constexpr int Bn = 4;
constexpr int CC = 512;    // C
constexpr int CQ = 64;     // CQK
constexpr int Hh = 128;
constexpr int Wd = 128;

typedef __attribute__((ext_vector_type(16))) __bf16 v16bf;
typedef __attribute__((ext_vector_type(8)))  float  v8f;

union Frag16 {            // 16 bf16 = 32 B, loaded as two 16-B chunks
    uint4 q[2];
    v16bf v;
};

// A-fragment (16-bit A 16x32 layout, ISA 7.12.2):
//   elements 0..7  = K[base + 8*lhi + 0..7]
//   elements 8..15 = K[base + 16 + 8*lhi + 0..7]
__device__ __forceinline__ v16bf loadA(const __bf16* rowbase, int lhi) {
    Frag16 f;
    f.q[0] = *(const uint4*)(rowbase + 8 * lhi);
    f.q[1] = *(const uint4*)(rowbase + 16 + 8 * lhi);
    return f.v;
}

// B-fragment (same K order as A):
//   elements 0..7  = K[base + 8*lhi + 0..7]
//   elements 8..15 = K[base + 16 + 8*lhi + 0..7]
__device__ __forceinline__ v16bf loadB(const __bf16* rowbase, int lhi) {
    Frag16 f;
    const __bf16* p = rowbase + 8 * lhi;
    f.q[0] = *(const uint4*)(p);
    f.q[1] = *(const uint4*)(p + 16);
    return f.v;
}

// Async global -> LDS 16-byte copy (GV mode), tracked by ASYNCcnt.
__device__ __forceinline__ void async_copy_b128(unsigned lds_byte_addr,
                                                const void* gaddr) {
    asm volatile("global_load_async_to_lds_b128 %0, %1, off"
                 :: "v"(lds_byte_addr),
                    "v"((unsigned long long)(uintptr_t)gaddr)
                 : "memory");
}
__device__ __forceinline__ void wait_async0() {
    asm volatile("s_wait_asynccnt 0x0" ::: "memory");
}

// ---------------------------------------------------------------------------
// Kernel 1: out[n, o] = sum_c x[n, c] * Wcat[o, c] + bcat[o]
//   n = ((b*H + h)*W + w), 16-wide tiles along w (contiguous x loads).
//   o in [0,640): [0,64) -> q, [64,128) -> k, [128,640) -> v.
// Each wave keeps 10 accumulator fragments so x is staged once per K-step.
// q/k stored [b][w][h][c]; v stored [b][w][c][h] (kernel 2's V tile is then
// K-contiguous for vectorized B-fragment loads).
// ---------------------------------------------------------------------------
__global__ __launch_bounds__(128) void qkv_proj_kernel(
    const float* __restrict__ x,
    const float* __restrict__ Wq, const float* __restrict__ bq,
    const float* __restrict__ Wk, const float* __restrict__ bk,
    const float* __restrict__ Wv, const float* __restrict__ bv,
    __bf16* __restrict__ q_ws, __bf16* __restrict__ k_ws,
    __bf16* __restrict__ v_ws)
{
    __shared__ __bf16 lds_a[16][40];     // [n-row][c]   rows 80 B
    __shared__ __bf16 lds_b[640][40];    // [o][c]       rows 80 B

    const int tid  = threadIdx.x;
    const int wv   = tid >> 5;
    const int lane = tid & 31;
    const int lhi  = lane >> 4;
    const int llo  = lane & 15;

    const int n0  = blockIdx.x * 16;
    const int b   = n0 / (Hh * Wd);
    const int rem = n0 % (Hh * Wd);
    const int h   = rem / Wd;
    const int w0  = rem % Wd;            // tile spans w0..w0+15 (W%16==0)

    const size_t xbase = (size_t)b * CC * Hh * Wd + (size_t)h * Wd + w0;

    v8f zero;
#pragma unroll
    for (int e = 0; e < 8; ++e) zero[e] = 0.0f;
    v8f acc[10];
#pragma unroll
    for (int p = 0; p < 10; ++p) acc[p] = zero;

    const int aj = tid & 15;             // x-transpose: this thread's n-row
    const int ac = (tid >> 4) * 4;       // and 4-channel chunk

    for (int kc = 0; kc < CC; kc += 32) {
        // Stage A: x tile 32c x 16n -> lds_a[n][c] (packed uint2 stores).
        {
            __bf16 t4[4];
#pragma unroll
            for (int i = 0; i < 4; ++i)
                t4[i] = (__bf16)x[xbase + (size_t)(kc + ac + i) * Hh * Wd + aj];
            *(uint2*)(&lds_a[aj][ac]) = *(const uint2*)t4;
        }
        // Stage B: 640 weight rows, 32 channels each -> lds_b[o][c].
#pragma unroll
        for (int s = 0; s < 5; ++s) {
            int og = tid + 128 * s;      // 0..639
            const float* wsrc; int orow;
            if (og < 64)        { wsrc = Wq; orow = og;       }
            else if (og < 128)  { wsrc = Wk; orow = og - 64;  }
            else                { wsrc = Wv; orow = og - 128; }
            const float4* rp = (const float4*)(wsrc + (size_t)orow * CC + kc);
            __bf16 t32[32];
#pragma unroll
            for (int i = 0; i < 8; ++i) {
                float4 f = rp[i];
                t32[4*i+0] = (__bf16)f.x; t32[4*i+1] = (__bf16)f.y;
                t32[4*i+2] = (__bf16)f.z; t32[4*i+3] = (__bf16)f.w;
            }
            uint4* dst = (uint4*)(&lds_b[og][0]);
#pragma unroll
            for (int i = 0; i < 4; ++i) dst[i] = ((const uint4*)t32)[i];
        }
        __syncthreads();

        v16bf a = loadA(&lds_a[llo][0], lhi);
#pragma unroll
        for (int p = 0; p < 10; ++p) {
            int oc = p * 64 + wv * 16 + llo;       // B column (N) row in LDS
            v16bf bm = loadB(&lds_b[oc][0], lhi);
            acc[p] = __builtin_amdgcn_wmma_f32_16x16x32_bf16(
                false, a, false, bm, (short)0, acc[p], false, false);
        }
        __syncthreads();
    }

    // Epilogue: add bias, store bf16 workspace.
#pragma unroll
    for (int p = 0; p < 10; ++p) {
        int og = p * 64 + wv * 16 + llo;
        if (og < 64) {
            float bias = bq[og];
#pragma unroll
            for (int r = 0; r < 8; ++r) {
                int wcol = w0 + r + 8 * lhi;
                q_ws[(((size_t)b * Wd + wcol) * Hh + h) * CQ + og] =
                    (__bf16)(acc[p][r] + bias);
            }
        } else if (og < 128) {
            int oo = og - 64; float bias = bk[oo];
#pragma unroll
            for (int r = 0; r < 8; ++r) {
                int wcol = w0 + r + 8 * lhi;
                k_ws[(((size_t)b * Wd + wcol) * Hh + h) * CQ + oo] =
                    (__bf16)(acc[p][r] + bias);
            }
        } else {
            int oo = og - 128; float bias = bv[oo];
#pragma unroll
            for (int r = 0; r < 8; ++r) {
                int wcol = w0 + r + 8 * lhi;
                // v layout: [b][w][c][h]
                v_ws[(((size_t)b * Wd + wcol) * CC + oo) * Hh + h] =
                    (__bf16)(acc[p][r] + bias);
            }
        }
    }
}

// ---------------------------------------------------------------------------
// Kernel 2: one block per (b,w). 256 threads = 8 waves.
//   S[h,g] = sum_c Q[h,c] K[g,c]     P = softmax_g(S)
//   O[h,c] = sum_g P[h,g] V[c][g]    out = gamma*O + x
// V tiles double-buffered via global_load_async_to_lds_b128 / s_wait_asynccnt.
// ---------------------------------------------------------------------------
__global__ __launch_bounds__(256) void axial_attn_kernel(
    const float*  __restrict__ x,
    const __bf16* __restrict__ q_ws,
    const __bf16* __restrict__ k_ws,
    const __bf16* __restrict__ v_ws,
    const float*  __restrict__ gamma,
    float*        __restrict__ out)
{
    extern __shared__ unsigned char smem[];
    __bf16* Qs = (__bf16*)smem;              // [128][72]  rows 144 B
    __bf16* Ks = Qs + 128 * 72;              // [128][72]
    __bf16* Ps = Ks + 128 * 72;              // [128][136] rows 272 B
    float*  S  = (float*)(Ps + 128 * 136);   // [128][132] rows 528 B
    __bf16* Vt = (__bf16*)(S + 128 * 132);   // [2][64][136]  V tiles: [c][g]

    const int tid  = threadIdx.x;
    const int wv   = tid >> 5;
    const int lane = tid & 31;
    const int lhi  = lane >> 4;
    const int llo  = lane & 15;

    const int bw = blockIdx.x;               // 0..511
    const int b  = bw / Wd;
    const int w  = bw % Wd;

    const unsigned vt_base = (unsigned)(uintptr_t)(void*)Vt;   // LDS byte addr
    const __bf16*  vcol    = v_ws + (size_t)bw * CC * Hh;      // this column's V

    // This thread's share of a 64x128 V tile: 4x 16-B transfers.
    int vrow[4], vseg[4];
#pragma unroll
    for (int s = 0; s < 4; ++s) {
        int vi = tid + 256 * s;              // 1024 uint4's = 64x128 bf16
        vrow[s] = vi >> 4;
        vseg[s] = vi & 15;
    }

    // ---- Phase 1: Q,K -> LDS (16-B vector copies) ----
    const uint4* qsrc = (const uint4*)(q_ws + (size_t)bw * Hh * CQ);
    const uint4* ksrc = (const uint4*)(k_ws + (size_t)bw * Hh * CQ);
#pragma unroll
    for (int s = 0; s < 4; ++s) {
        int vi  = tid + 256 * s;             // 1024 uint4's total
        int hr  = vi >> 3, seg = vi & 7;
        *(uint4*)(Qs + hr * 72 + seg * 8) = qsrc[vi];
        *(uint4*)(Ks + hr * 72 + seg * 8) = ksrc[vi];
    }
    __syncthreads();

    // Kick off async load of V chunk 0 into buffer 0 (overlaps QK^T+softmax).
#pragma unroll
    for (int s = 0; s < 4; ++s) {
        unsigned lds_addr = vt_base + (unsigned)((vrow[s] * 136 + vseg[s] * 8) * 2);
        async_copy_b128(lds_addr, vcol + (size_t)vrow[s] * Hh + vseg[s] * 8);
    }

    // ---- Phase 2: S = Q K^T. Wave wv owns h-tile wv, all 8 g-tiles. ----
#pragma unroll
    for (int j = 0; j < 8; ++j) {
        v8f acc;
#pragma unroll
        for (int e = 0; e < 8; ++e) acc[e] = 0.0f;
#pragma unroll
        for (int kc = 0; kc < CQ; kc += 32) {
            v16bf a  = loadA(Qs + (wv * 16 + llo) * 72 + kc, lhi);
            v16bf bm = loadB(Ks + (j  * 16 + llo) * 72 + kc, lhi);
            acc = __builtin_amdgcn_wmma_f32_16x16x32_bf16(
                false, a, false, bm, (short)0, acc, false, false);
        }
#pragma unroll
        for (int r = 0; r < 8; ++r)
            S[(wv * 16 + r + 8 * lhi) * 132 + j * 16 + llo] = acc[r];
    }
    __syncthreads();

    // ---- Softmax over g, one row per thread (tid < 128). ----
    if (tid < 128) {
        float* row = S + tid * 132;
        float4* row4 = (float4*)row;
        float m = -3.0e38f;
#pragma unroll 8
        for (int g4 = 0; g4 < 32; ++g4) {
            float4 f = row4[g4];
            m = fmaxf(m, fmaxf(fmaxf(f.x, f.y), fmaxf(f.z, f.w)));
        }
        float sum = 0.0f;
#pragma unroll 8
        for (int g4 = 0; g4 < 32; ++g4) {
            float4 f = row4[g4];
            f.x = __expf(f.x - m); f.y = __expf(f.y - m);
            f.z = __expf(f.z - m); f.w = __expf(f.w - m);
            sum += f.x + f.y + f.z + f.w;
            row4[g4] = f;
        }
        float inv = 1.0f / sum;
        __bf16* prow = Ps + tid * 136;
#pragma unroll 4
        for (int g8 = 0; g8 < 16; ++g8) {     // pack 8 bf16 -> uint4 store
            float4 f0 = row4[2 * g8], f1 = row4[2 * g8 + 1];
            __bf16 pb[8];
            pb[0] = (__bf16)(f0.x * inv); pb[1] = (__bf16)(f0.y * inv);
            pb[2] = (__bf16)(f0.z * inv); pb[3] = (__bf16)(f0.w * inv);
            pb[4] = (__bf16)(f1.x * inv); pb[5] = (__bf16)(f1.y * inv);
            pb[6] = (__bf16)(f1.z * inv); pb[7] = (__bf16)(f1.w * inv);
            *(uint4*)(prow + g8 * 8) = *(const uint4*)pb;
        }
    }

    const float g0 = gamma[0];

    // ---- Phase 3: O = P * V, double-buffered async V tiles. ----
    for (int it = 0; it < 8; ++it) {
        const int c0  = it * 64;
        const int buf = it & 1;

        // Wait for this iteration's V tile (issued last iteration), then make
        // it (and the softmax probs on iter 0) visible to all waves.
        wait_async0();
        __syncthreads();

        // Prefetch next chunk into the other buffer while we compute.
        if (it + 1 < 8) {
            const int nbuf = buf ^ 1;
            const int c1   = c0 + 64;
#pragma unroll
            for (int s = 0; s < 4; ++s) {
                unsigned lds_addr = vt_base +
                    (unsigned)(((nbuf * 64 + vrow[s]) * 136 + vseg[s] * 8) * 2);
                async_copy_b128(lds_addr,
                                vcol + (size_t)(c1 + vrow[s]) * Hh + vseg[s] * 8);
            }
        }

        const __bf16* Vb = Vt + buf * 64 * 136;
#pragma unroll
        for (int jj = 0; jj < 4; ++jj) {     // wave wv: h-tile wv, 4 c-subtiles
            v8f acc;
#pragma unroll
            for (int e = 0; e < 8; ++e) acc[e] = 0.0f;
#pragma unroll
            for (int g8 = 0; g8 < 128; g8 += 32) {
                v16bf a  = loadA(Ps + (wv * 16 + llo) * 136 + g8, lhi);
                v16bf bm = loadB(Vb + (jj * 16 + llo) * 136 + g8, lhi);
                acc = __builtin_amdgcn_wmma_f32_16x16x32_bf16(
                    false, a, false, bm, (short)0, acc, false, false);
            }
            int cg = c0 + jj * 16 + llo;
#pragma unroll
            for (int r = 0; r < 8; ++r) {
                int hh = wv * 16 + r + 8 * lhi;
                size_t idx = (((size_t)b * CC + cg) * Hh + hh) * Wd + w;
                out[idx] = g0 * acc[r] + x[idx];
            }
        }
    }
}

// ---------------------------------------------------------------------------
// Host launcher. Workspace layout (bf16):
//   q_ws [B][W][H][64]   = 8 MB
//   k_ws [B][W][H][64]   = 8 MB
//   v_ws [B][W][512][H]  = 64 MB       (total 80 MB)
// ---------------------------------------------------------------------------
extern "C" void kernel_launch(void* const* d_in, const int* in_sizes, int n_in,
                              void* d_out, int out_size, void* d_ws, size_t ws_size,
                              hipStream_t stream) {
    const float* x     = (const float*)d_in[0];
    const float* Wq    = (const float*)d_in[1];
    const float* bq    = (const float*)d_in[2];
    const float* Wk    = (const float*)d_in[3];
    const float* bk    = (const float*)d_in[4];
    const float* Wv    = (const float*)d_in[5];
    const float* bv    = (const float*)d_in[6];
    const float* gamma = (const float*)d_in[7];

    __bf16* q_ws = (__bf16*)d_ws;
    __bf16* k_ws = q_ws + (size_t)Bn * Wd * Hh * CQ;
    __bf16* v_ws = k_ws + (size_t)Bn * Wd * Hh * CQ;

    // Kernel 1: 65536/16 = 4096 n-tiles, 128 threads (4 waves) each.
    qkv_proj_kernel<<<dim3(4096), dim3(128), 0, stream>>>(
        x, Wq, bq, Wk, bk, Wv, bv, q_ws, k_ws, v_ws);

    // Kernel 2: one block per (b,w) column; ~170 KB dynamic LDS.
    size_t smem_bytes = (size_t)128 * 72 * 2        // Qs
                      + (size_t)128 * 72 * 2        // Ks
                      + (size_t)128 * 136 * 2       // Ps
                      + (size_t)128 * 132 * 4       // S
                      + (size_t)2 * 64 * 136 * 2;   // Vt (double buffer)
    axial_attn_kernel<<<dim3(Bn * Wd), dim3(256), smem_bytes, stream>>>(
        x, q_ws, k_ws, v_ws, gamma, (float*)d_out);
}
